// SC_Dec_24077586662016
// MI455X (gfx1250) — 8-bit matrix operand order repaired and checked by reading, compiled, not run on hardware
//
#include <hip/hip_runtime.h>
#include <hip/hip_bf16.h>
#include <stdint.h>

typedef __attribute__((ext_vector_type(8))) int v8i;

#define NCODE 256
#define HALF  128
#define ROWS_PER_BLOCK 128
#define THREADS 256

__global__ __launch_bounds__(THREADS)
void polar_sc_wmma_kernel(const float* __restrict__ in,
                          float* __restrict__ out,
                          int Btotal) {
  __shared__ unsigned char beta[ROWS_PER_BLOCK * HALF];
  __shared__ unsigned char gt[HALF * HALF];

  const int tid  = threadIdx.x;
  const int b0   = blockIdx.x * ROWS_PER_BLOCK;
  const bool full = (b0 + ROWS_PER_BLOCK) <= Btotal;

  #pragma unroll 4
  for (int i = 0; i < (HALF * HALF) / THREADS; ++i) {
    int idx = i * THREADS + tid;
    int n = idx >> 7;
    int k = idx & (HALF - 1);
    gt[idx] = (unsigned char)(((k & n) == n) ? 1 : 0);
  }

  const float* inb = in + (size_t)b0 * NCODE;
  if (full) {
    #pragma unroll 4
    for (int i = 0; i < (ROWS_PER_BLOCK * HALF) / THREADS; ++i) {
      int idx = i * THREADS + tid;
      int b = idx >> 7;
      int j = idx & (HALF - 1);
      float s = inb[b * NCODE + j] + inb[b * NCODE + j + HALF];
      beta[idx] = (s >= 0.0f) ? (unsigned char)1 : (unsigned char)0;
    }
  } else {
    for (int i = 0; i < (ROWS_PER_BLOCK * HALF) / THREADS; ++i) {
      int idx = i * THREADS + tid;
      int b = idx >> 7;
      int j = idx & (HALF - 1);
      unsigned char bit = 0;
      if (b0 + b < Btotal) {
        float s = inb[b * NCODE + j] + inb[b * NCODE + j + HALF];
        bit = (s >= 0.0f) ? (unsigned char)1 : (unsigned char)0;
      }
      beta[idx] = bit;
    }
  }
  __syncthreads();

  const int wave = tid >> 5;
  const int lane = tid & 31;
  const int half = lane >> 4;
  const int mr   = lane & 15;
  const int m0   = wave * 16;

  v8i a0, a1;
  #pragma unroll
  for (int v = 0; v < 8; ++v) {
    int K0 = ((v >> 2) << 5) | (half << 4) | ((v & 3) << 2);
    const int rowoff = (m0 + mr) * HALF;
    a0[v] = (int)*(const unsigned int*)&beta[rowoff + K0];
    a1[v] = (int)*(const unsigned int*)&beta[rowoff + 64 + K0];
  }
  __syncthreads();

  #pragma unroll
  for (int nt = 0; nt < 8; ++nt) {
    const int ncol = nt * 16 + mr;
    v8i bf0, bf1;
    #pragma unroll
    for (int v = 0; v < 8; ++v) {
      int Kb = ((v >> 2) << 5) | (half << 4) | ((v & 3) << 2);
      const int rowoff = ncol * HALF;
      bf0[v] = (int)*(const unsigned int*)&gt[rowoff + Kb];
      bf1[v] = (int)*(const unsigned int*)&gt[rowoff + 64 + Kb];
    }
    v8i c = {0, 0, 0, 0, 0, 0, 0, 0};
    c = __builtin_amdgcn_wmma_i32_16x16x64_iu8(false, a0, false, bf0, c, false, false);
    c = __builtin_amdgcn_wmma_i32_16x16x64_iu8(false, a1, false, bf1, c, false, false);

    #pragma unroll
    for (int r = 0; r < 8; ++r) {
      int mrow = m0 + half * 8 + r;
      beta[mrow * HALF + ncol] = (unsigned char)(c[r] & 1);
    }
  }
  __syncthreads();

  float* outb = out + (size_t)b0 * HALF;
  if (full) {
    #pragma unroll 4
    for (int i = 0; i < (ROWS_PER_BLOCK * HALF) / THREADS; ++i) {
      int idx = i * THREADS + tid;
      outb[idx] = (float)beta[idx];
    }
  } else {
    for (int i = 0; i < (ROWS_PER_BLOCK * HALF) / THREADS; ++i) {
      int idx = i * THREADS + tid;
      if (b0 + (idx >> 7) < Btotal) outb[idx] = (float)beta[idx];
    }
  }
}

extern "C" void kernel_launch(void* const* d_in, const int* in_sizes, int n_in,
                              void* d_out, int out_size, void* d_ws, size_t ws_size,
                              hipStream_t stream) {
  const float* in = (const float*)d_in[0];
  float* out = (float*)d_out;
  const int Btotal = in_sizes[0] / NCODE;
  const int blocks = (Btotal + ROWS_PER_BLOCK - 1) / ROWS_PER_BLOCK;
  polar_sc_wmma_kernel<<<blocks, THREADS, 0, stream>>>(in, out, Btotal);
}
